// ASAP_Pool_88330297410222
// MI455X (gfx1250) — matrix operand order repaired and checked by reading, not compiled, not run on hardware
//
#include <hip/hip_runtime.h>

// ---------------------------------------------------------------------------
// Types for CDNA5 WMMA (wave32)
// ---------------------------------------------------------------------------
typedef __attribute__((ext_vector_type(16))) _Float16 v16h;
typedef __attribute__((ext_vector_type(8)))  _Float16 v8h;
typedef __attribute__((ext_vector_type(8)))  float    v8f;

#define BIG_NEG (-1e9f)
#define LEAK 0.2f

// ---------------------------------------------------------------------------
// WMMA GEMM: D[b] = A[b] (MxK f16 row-major) x B (KxN via BT = NxK row-major f16)
// f32 accumulate, optional per-column bias + relu epilogue.
//
// Workgroup = 4 waves, covers a 16-row x 256-col output strip.
// The 16 x Kp A strip is staged into (dynamically sized) LDS once via
// global_load_async_to_lds_b128 (ASYNCcnt) and then read as ds_load_b128
// fragments; each wave computes 4 x 16x16 tiles with explicit double
// buffering: K-step k+32 fragments are loaded before the 4 WMMAs of step k.
// K padded to 32; lda/ldb multiples of 32 elements; Kp <= 1024.
// ---------------------------------------------------------------------------
__global__ void __launch_bounds__(128)
wmma_gemm_f16(const _Float16* __restrict__ A, const _Float16* __restrict__ BT,
              float* __restrict__ D,
              int Mt, int Nt, int Kp, int lda, int ldb, int ldc,
              int Mvalid, int Nvalid,
              long sA, long sB, long sD,
              const float* __restrict__ bias, int relu)
{
    extern __shared__ _Float16 shA[];            // 16 x Kp strip (launch-sized)
    int bz   = blockIdx.y;
    int nWGn = (Nt + 15) >> 4;                   // workgroups along N
    int tm   = blockIdx.x / nWGn;
    int wgn  = blockIdx.x % nWGn;
    int wave = threadIdx.x >> 5;
    int lane = threadIdx.x & 31;

    // ---- async-copy the 16 x Kp A strip into LDS (128 lanes x 16B chunks) ----
    const _Float16* Abase = A + bz * sA + (long)(tm * 16) * lda;
    int cpk    = Kp >> 3;                        // 16-byte chunks per row
    int chunks = cpk * 16;
    for (int c = threadIdx.x; c < chunks; c += 128) {
        int row = c / cpk, col = (c - row * cpk) << 3;
        unsigned lds_off = (unsigned)(size_t)(shA + row * Kp + col);
        const _Float16* g = Abase + (long)row * lda + col;
        asm volatile("global_load_async_to_lds_b128 %0, %1, off"
                     :: "v"(lds_off), "v"(g) : "memory");
    }
    asm volatile("s_wait_asynccnt 0x0" ::: "memory");
    __syncthreads();

    // ---- per-wave: 4 consecutive 16-col tiles ----
    int tn0   = wgn * 16 + wave * 4;
    int m     = lane & 15;
    int abase = (lane >> 4) << 3;                // D row offset 0 / 8; A frag K offset 2x = 0 / 16
    int boff  = (lane >> 4) << 4;                // B frag K offset: 0 / 16

    const _Float16* Bg = BT + bz * sB;
    int t0 = tn0 + 0, t1 = tn0 + 1, t2 = tn0 + 2, t3 = tn0 + 3;
    int c0 = t0 < Nt ? t0 : Nt - 1;
    int c1 = t1 < Nt ? t1 : Nt - 1;
    int c2 = t2 < Nt ? t2 : Nt - 1;
    int c3 = t3 < Nt ? t3 : Nt - 1;
    const _Float16* Bp0 = Bg + (long)(c0 * 16 + m) * ldb + boff;
    const _Float16* Bp1 = Bg + (long)(c1 * 16 + m) * ldb + boff;
    const _Float16* Bp2 = Bg + (long)(c2 * 16 + m) * ldb + boff;
    const _Float16* Bp3 = Bg + (long)(c3 * 16 + m) * ldb + boff;
    const _Float16* As  = shA + m * Kp + 2 * abase;

    // ---- double-buffered main loop ----
    int kLast = Kp - 32;
    v8h  a0 = *(const v8h*)(As);
    v8h  a1 = *(const v8h*)(As + 8);
    v16h b0 = *(const v16h*)(Bp0);
    v16h b1 = *(const v16h*)(Bp1);
    v16h b2 = *(const v16h*)(Bp2);
    v16h b3 = *(const v16h*)(Bp3);

    v8f acc0 = {}, acc1 = {}, acc2 = {}, acc3 = {};
    for (int k0 = 0; k0 < Kp; k0 += 32) {
        int kn = (k0 + 32 <= kLast) ? k0 + 32 : kLast;   // clamped next (branch-free)
        v8h  na0 = *(const v8h*)(As + kn);               // ds_load_b128
        v8h  na1 = *(const v8h*)(As + kn + 8);
        v16h nb0 = *(const v16h*)(Bp0 + kn);             // global_load_b128 x2 each
        v16h nb1 = *(const v16h*)(Bp1 + kn);
        v16h nb2 = *(const v16h*)(Bp2 + kn);
        v16h nb3 = *(const v16h*)(Bp3 + kn);
        if (k0 + 256 < Kp) {
            __builtin_prefetch((const void*)(Bp0 + k0 + 256), 0, 1);
            __builtin_prefetch((const void*)(Bp1 + k0 + 256), 0, 1);
            __builtin_prefetch((const void*)(Bp2 + k0 + 256), 0, 1);
            __builtin_prefetch((const void*)(Bp3 + k0 + 256), 0, 1);
        }
        v16h av;
#pragma unroll
        for (int e = 0; e < 8; ++e) { av[e] = a0[e]; av[e + 8] = a1[e]; }
        acc0 = __builtin_amdgcn_wmma_f32_16x16x32_f16(false, av, false, b0, (short)0, acc0, false, false);
        acc1 = __builtin_amdgcn_wmma_f32_16x16x32_f16(false, av, false, b1, (short)0, acc1, false, false);
        acc2 = __builtin_amdgcn_wmma_f32_16x16x32_f16(false, av, false, b2, (short)0, acc2, false, false);
        acc3 = __builtin_amdgcn_wmma_f32_16x16x32_f16(false, av, false, b3, (short)0, acc3, false, false);
        a0 = na0; a1 = na1; b0 = nb0; b1 = nb1; b2 = nb2; b3 = nb3;
    }

    // ---- epilogue: bias + relu, guarded stores (no duplicate writes) ----
    int rbase = tm * 16 + abase;
    float* Dp = D + bz * sD;
#define STORE_TILE(TIDX, ACC)                                              \
    if ((TIDX) < Nt) {                                                     \
        int col = (TIDX) * 16 + m;                                         \
        if (col < Nvalid) {                                                \
            float bs = bias ? bias[col] : 0.0f;                            \
            _Pragma("unroll")                                              \
            for (int r = 0; r < 8; ++r) {                                  \
                int row = rbase + r;                                       \
                if (row < Mvalid) {                                        \
                    float v = ACC[r] + bs;                                 \
                    if (relu) v = v > 0.f ? v : 0.f;                       \
                    Dp[(long)row * ldc + col] = v;                         \
                }                                                          \
            }                                                              \
        }                                                                  \
    }
    STORE_TILE(t0, acc0)
    STORE_TILE(t1, acc1)
    STORE_TILE(t2, acc2)
    STORE_TILE(t3, acc3)
#undef STORE_TILE
}

// ---------------------------------------------------------------------------
// f32 -> f16 convert with zero padding (A operands).  out is [Rp x Cp] row-major.
// ---------------------------------------------------------------------------
__global__ void convert_pad_f16(const float* __restrict__ in, _Float16* __restrict__ out,
                                int R, int Ccols, int ldin, int Rp, int Cp,
                                long sIn, long sOut)
{
    int b = blockIdx.y;
    int idx = blockIdx.x * 256 + threadIdx.x;
    if (idx >= Rp * Cp) return;
    int r = idx / Cp, c = idx % Cp;
    float v = (r < R && c < Ccols) ? in[b * sIn + (long)r * ldin + c] : 0.f;
    out[b * sOut + idx] = (_Float16)v;
}

// f32 [R x Ccols] -> f16 transposed [pad16(Ccols) x pad32(R)] zero padded (BT operands)
__global__ void transpose_convert_f16(const float* __restrict__ in, _Float16* __restrict__ out,
                                      int R, int Ccols, int ldin, int Rout, int Cout,
                                      long sIn, long sOut)
{
    int b = blockIdx.y;
    int idx = blockIdx.x * 256 + threadIdx.x;
    if (idx >= Rout * Cout) return;
    int ro = idx / Cout, co = idx % Cout;
    float v = (ro < Ccols && co < R) ? in[b * sIn + (long)co * ldin + ro] : 0.f;
    out[b * sOut + idx] = (_Float16)v;
}

// ---------------------------------------------------------------------------
// y[b,i] = dot(X[b,i,0:512], w)      one wave (32 lanes) per row
// ---------------------------------------------------------------------------
__global__ void gemv_dot(const float* __restrict__ X, int ldx, long sX,
                         const float* __restrict__ w, float* __restrict__ y,
                         int n, int total, int Clen)
{
    int id = blockIdx.x * 4 + (threadIdx.x >> 5);
    if (id >= total) return;
    int lane = threadIdx.x & 31;
    int b = id / n, i = id % n;
    const float* xp = X + (long)b * sX + (long)i * ldx;
    float s = 0.f;
    for (int c = lane * 4; c < Clen; c += 128) {
        float4 v = *(const float4*)(xp + c);
        s += v.x * w[c] + v.y * w[c + 1] + v.z * w[c + 2] + v.w * w[c + 3];
    }
    for (int o = 16; o; o >>= 1) s += __shfl_down(s, o, 32);
    if (!lane) y[(long)b * 256 + i] = s;
}

// deg[b,i] = sum_j A[b,i,j]
__global__ void rowsum_kernel(const float* __restrict__ A, float* __restrict__ deg,
                              int n, int total)
{
    int id = blockIdx.x * 4 + (threadIdx.x >> 5);
    if (id >= total) return;
    int lane = threadIdx.x & 31;
    int b = id / n, i = id % n;
    const float* row = A + ((long)b * 256 + i) * 256;
    float s = 0.f;
    for (int j = lane; j < n; j += 32) s += row[j];
    for (int o = 16; o; o >>= 1) s += __shfl_down(s, o, 32);
    if (!lane) deg[(long)b * 256 + i] = s;
}

// ---------------------------------------------------------------------------
// masked softmax row: logit = mask ? leaky(si_i + sj_j + eb) : BIG_NEG
// writes f16 [pad16 rows x npc cols] zero-padded.  grid=(np, B), block=256
// ---------------------------------------------------------------------------
__global__ void masked_softmax_f16(const float* __restrict__ A,
                                   const float* __restrict__ si, const float* __restrict__ sj,
                                   const float* __restrict__ ebp,
                                   int n, int npc, _Float16* __restrict__ out, long sOut)
{
    int b = blockIdx.y, i = blockIdx.x, j = threadIdx.x;
    __shared__ float red[256];
    float eb = ebp ? ebp[0] : 0.f;
    bool in_range = (i < n) && (j < n);
    float v = BIG_NEG;
    if (in_range && A[((long)b * 256 + i) * 256 + j] > 0.f) {
        float t = si[(long)b * 256 + i] + sj[(long)b * 256 + j] + eb;
        v = t > 0.f ? t : LEAK * t;
    }
    red[j] = v; __syncthreads();
    for (int s = 128; s; s >>= 1) { if (j < s) red[j] = fmaxf(red[j], red[j + s]); __syncthreads(); }
    float mx = red[0]; __syncthreads();
    float e = in_range ? __expf(v - mx) : 0.f;   // exp(BIG_NEG-mx) underflows to 0 like reference
    red[j] = e; __syncthreads();
    for (int s = 128; s; s >>= 1) { if (j < s) red[j] += red[j + s]; __syncthreads(); }
    float sum = red[0];
    float r = (sum > 0.f) ? e / sum : 0.f;
    if (j < npc) out[b * sOut + (long)i * npc + j] = (_Float16)r;
}

// Â = A * d_i * d_j (sym-norm GCN adjacency) written as f16 padded operand
__global__ void norm_adj_f16(const float* __restrict__ A, const float* __restrict__ deg,
                             _Float16* __restrict__ out, int n, int npc, long sOut)
{
    int b = blockIdx.y, i = blockIdx.x, j = threadIdx.x;
    if (j >= npc) return;
    float v = 0.f;
    if (i < n && j < n) {
        float a = A[((long)b * 256 + i) * 256 + j];
        float di = deg[(long)b * 256 + i];
        float dj = deg[(long)b * 256 + j];
        di = di > 0.f ? rsqrtf(di) : 0.f;
        dj = dj > 0.f ? rsqrtf(dj) : 0.f;
        v = a * di * dj;
    }
    out[b * sOut + (long)i * npc + j] = (_Float16)v;
}

// Xq[b,i,c] = max_{j: A[b,i,j]>0} xp[b,j,c]  (init BIG_NEG)
__global__ void neighbor_max_kernel(const float* __restrict__ A, const float* __restrict__ xp,
                                    float* __restrict__ Xq, int n)
{
    int b = blockIdx.y, i = blockIdx.x, t = threadIdx.x;   // 128 threads
    __shared__ float flags[256];
    for (int j = t; j < n; j += 128) flags[j] = A[((long)b * 256 + i) * 256 + j];
    __syncthreads();
    float a0 = BIG_NEG, a1 = BIG_NEG, a2 = BIG_NEG, a3 = BIG_NEG;
    for (int j = 0; j < n; ++j) {
        if (flags[j] > 0.f) {
            const float* xr = xp + ((long)b * 256 + j) * 512;
            a0 = fmaxf(a0, xr[t]);
            a1 = fmaxf(a1, xr[t + 128]);
            a2 = fmaxf(a2, xr[t + 256]);
            a3 = fmaxf(a3, xr[t + 384]);
        }
    }
    float* o = Xq + ((long)b * 256 + i) * 512;
    o[t] = a0; o[t + 128] = a1; o[t + 256] = a2; o[t + 384] = a3;
}

// fit = sigmoid(t1 + b1 + t2*deg - sum_j A[i,j]*t3[j])
__global__ void fit_kernel(const float* __restrict__ A, const float* __restrict__ t1,
                           const float* __restrict__ t2, const float* __restrict__ t3,
                           const float* __restrict__ deg, const float* __restrict__ b1p,
                           float* __restrict__ fit, int n)
{
    int b = blockIdx.y, i = blockIdx.x, t = threadIdx.x;
    __shared__ float red[256];
    const float* row = A + ((long)b * 256 + i) * 256;
    float s = 0.f;
    for (int j = t; j < n; j += 256) s += row[j] * t3[(long)b * 256 + j];
    red[t] = s; __syncthreads();
    for (int st = 128; st; st >>= 1) { if (t < st) red[t] += red[t + st]; __syncthreads(); }
    if (!t) {
        long o = (long)b * 256 + i;
        float z = t1[o] + b1p[0] + t2[o] * deg[o] - red[0];
        fit[o] = 1.f / (1.f + __expf(-z));
    }
}

// top-k with ties -> lowest index (matches lax.top_k), one block per graph
__global__ void topk_kernel(const float* __restrict__ fit, int n, int k,
                            int* __restrict__ perm, float* __restrict__ vals)
{
    int b = blockIdx.x, t = threadIdx.x;
    __shared__ float mv[256]; __shared__ float sv[256]; __shared__ int sidx[256];
    mv[t] = (t < n) ? fit[(long)b * 256 + t] : -3.4e38f;
    __syncthreads();
    for (int it = 0; it < k; ++it) {
        sv[t] = mv[t]; sidx[t] = t; __syncthreads();
        for (int s = 128; s; s >>= 1) {
            if (t < s) {
                if (sv[t + s] > sv[t] || (sv[t + s] == sv[t] && sidx[t + s] < sidx[t])) {
                    sv[t] = sv[t + s]; sidx[t] = sidx[t + s];
                }
            }
            __syncthreads();
        }
        if (!t) { perm[(long)b * 256 + it] = sidx[0]; vals[(long)b * 256 + it] = sv[0]; mv[sidx[0]] = -3.4e38f; }
        __syncthreads();
    }
}

// x_next[b,t,:] = xc[b,perm[t],:] * vals[b,t]
__global__ void gather_scale_kernel(const float* __restrict__ xc, const int* __restrict__ perm,
                                    const float* __restrict__ vals, float* __restrict__ xo)
{
    int b = blockIdx.y, t = blockIdx.x, c = threadIdx.x;  // 128 threads x float4
    int p = perm[(long)b * 256 + t];
    float v = vals[(long)b * 256 + t];
    const float4* s = (const float4*)(xc + ((long)b * 256 + p) * 512);
    float4* d = (float4*)(xo + ((long)b * 256 + t) * 512);
    float4 q = s[c]; q.x *= v; q.y *= v; q.z *= v; q.w *= v; d[c] = q;
}

// Sk[b,t,:] = S[b,perm[t],:]  (f16, zero pad t>=k)
__global__ void gather_rows_f16(const _Float16* __restrict__ S, const int* __restrict__ perm,
                                _Float16* __restrict__ out, int k, int npc, long sS, long sOut)
{
    int b = blockIdx.y, t = blockIdx.x, j = threadIdx.x;
    if (j >= npc) return;
    _Float16 v = (_Float16)0.f;
    if (t < k) { int p = perm[(long)b * 256 + t]; v = S[b * sS + (long)p * npc + j]; }
    out[b * sOut + (long)t * npc + j] = v;
}

__global__ void diag_fix_kernel(float* __restrict__ A, int k)
{
    int b = blockIdx.x, i = threadIdx.x;
    if (i < k) {
        long o = ((long)b * 256 + i) * 256 + i;
        float v = A[o];
        if (v <= 0.f) A[o] = v + 1.f;
    }
}

// xs[b] += concat(mean_i x[b,i,:], max_i x[b,i,:])
__global__ void readout_kernel(const float* __restrict__ x, int k, float* __restrict__ xs)
{
    int b = blockIdx.x, t = threadIdx.x;  // 256 threads
    for (int c = t; c < 512; c += 256) {
        float sm = 0.f, mx = -3.4e38f;
        const float* xp = x + (long)b * 131072 + c;
        for (int i = 0; i < k; ++i) { float v = xp[(long)i * 512]; sm += v; mx = fmaxf(mx, v); }
        xs[(long)b * 1024 + c]       += sm / (float)k;
        xs[(long)b * 1024 + 512 + c] += mx;
    }
}

__global__ void zero_f32(float* __restrict__ p, int count)
{
    int i = blockIdx.x * 256 + threadIdx.x;
    if (i < count) p[i] = 0.f;
}

__global__ void embed_kernel(const int* __restrict__ ids, const float* __restrict__ emb,
                             float* __restrict__ x)
{
    int b = blockIdx.y, i = blockIdx.x, t = threadIdx.x;  // 128 threads x float4 = 512 floats
    int id = ids[(long)b * 256 + i];
    const float4* src = (const float4*)(emb + (long)id * 512);
    float4* dst = (float4*)(x + ((long)b * 256 + i) * 512);
    dst[t] = src[t];
}

__global__ void a_init_kernel(const float* __restrict__ adj, float* __restrict__ A)
{
    int idx = blockIdx.x * 256 + threadIdx.x;   // B*256*256 total
    int j = idx & 255, i = (idx >> 8) & 255;
    float v = adj[idx];
    if (i == j) v = fmaxf(v, 1.f);
    A[idx] = v;
}

// ---------------------------------------------------------------------------
// Host orchestration
// ---------------------------------------------------------------------------
extern "C" void kernel_launch(void* const* d_in, const int* in_sizes, int n_in,
                              void* d_out, int out_size, void* d_ws, size_t ws_size,
                              hipStream_t stream)
{
    (void)in_sizes; (void)n_in; (void)out_size; (void)ws_size;
    const int Bx = 256, Nn = 256, Cc = 512;

    const int*   x_ids  = (const int*)  d_in[0];
    const float* adj    = (const float*)d_in[2];
    const float* emb    = (const float*)d_in[3];
    const float* conv_W = (const float*)d_in[4];
    const float* conv_b = (const float*)d_in[5];
    const float* att_src= (const float*)d_in[6];
    const float* att_dst= (const float*)d_in[7];
    const float* q_W    = (const float*)d_in[8];
    const float* q_b    = (const float*)d_in[9];
    const float* att_w  = (const float*)d_in[10];
    const float* att_b  = (const float*)d_in[11];
    const float* gcn_W  = (const float*)d_in[12];
    const float* gcn_b  = (const float*)d_in[13];
    const float* le_W1  = (const float*)d_in[14];
    const float* le_b1  = (const float*)d_in[15];
    const float* le_W2  = (const float*)d_in[16];
    const float* le_W3  = (const float*)d_in[17];
    const float* lin1_W = (const float*)d_in[18];
    const float* lin1_b = (const float*)d_in[19];
    const float* lin2_W = (const float*)d_in[20];
    const float* lin2_b = (const float*)d_in[21];
    float* out = (float*)d_out;

    size_t cur = 0; char* wsb = (char*)d_ws;
    auto alloc = [&](size_t bytes) -> void* {
        void* p = wsb + cur; cur = (cur + bytes + 255) & ~(size_t)255; return p;
    };
    float* xF  = (float*)alloc((size_t)Bx * Nn * Cc * 4);
    float* t0F = (float*)alloc((size_t)Bx * Nn * Cc * 4);
    float* t1F = (float*)alloc((size_t)Bx * Nn * Cc * 4);
    float* t2F = (float*)alloc((size_t)Bx * Nn * Cc * 4);
    float* t3F = (float*)alloc((size_t)Bx * Nn * Cc * 4);
    float* A0  = (float*)alloc((size_t)Bx * Nn * Nn * 4);
    float* A1  = (float*)alloc((size_t)Bx * Nn * Nn * 4);
    float* SAF = (float*)alloc((size_t)Bx * Nn * Nn * 4);
    float* degF = (float*)alloc((size_t)Bx * 256 * 4);
    float* siF  = (float*)alloc((size_t)Bx * 256 * 4);
    float* sjF  = (float*)alloc((size_t)Bx * 256 * 4);
    float* v1F  = (float*)alloc((size_t)Bx * 256 * 4);
    float* v2F  = (float*)alloc((size_t)Bx * 256 * 4);
    float* v3F  = (float*)alloc((size_t)Bx * 256 * 4);
    float* fitF = (float*)alloc((size_t)Bx * 256 * 4);
    float* valsF= (float*)alloc((size_t)Bx * 256 * 4);
    int*   permI= (int*)  alloc((size_t)Bx * 256 * 4);
    float* xsF  = (float*)alloc((size_t)256 * 1024 * 4);
    float* h1F  = (float*)alloc((size_t)256 * 512 * 4);
    _Float16* bX  = (_Float16*)alloc((size_t)Bx * Nn * Cc * 2);
    _Float16* bT  = (_Float16*)alloc((size_t)Bx * Cc * Nn * 2);
    _Float16* bS  = (_Float16*)alloc((size_t)Bx * Nn * Nn * 2);
    _Float16* bSk = (_Float16*)alloc((size_t)Bx * Nn * Nn * 2);
    _Float16* bAb = (_Float16*)alloc((size_t)Bx * Nn * Nn * 2);
    _Float16* bW  = (_Float16*)alloc((size_t)1024 * 1024 * 2);

    auto pad16 = [](int v) { return (v + 15) & ~15; };
    auto pad32 = [](int v) { return (v + 31) & ~31; };

    auto gemm = [&](const _Float16* Aop, int lda, long sA,
                    const _Float16* Bop, int ldb, long sB,
                    float* D, int ldc, long sD,
                    int Mv, int Nv, int Kp, int Mt, int Nt, int batch,
                    const float* bias, int relu) {
        int nWGn = (Nt + 15) >> 4;
        dim3 grid(Mt * nWGn, batch, 1);
        size_t shbytes = (size_t)16 * Kp * sizeof(_Float16);
        wmma_gemm_f16<<<grid, 128, shbytes, stream>>>(Aop, Bop, D, Mt, Nt, Kp, lda, ldb, ldc,
                                                      Mv, Nv, sA, sB, sD, bias, relu);
    };
    auto conv = [&](const float* in, int ldin, long sIn, _Float16* o,
                    int R, int Cl, int Rp, int Cp, long sOut, int batch) {
        int per = Rp * Cp; dim3 grid((per + 255) / 256, batch, 1);
        convert_pad_f16<<<grid, 256, 0, stream>>>(in, o, R, Cl, ldin, Rp, Cp, sIn, sOut);
    };
    auto tconv = [&](const float* in, int ldin, long sIn, _Float16* o,
                     int R, int Cl, long sOut, int batch) {
        int Rout = pad16(Cl), Cout = pad32(R);
        int per = Rout * Cout; dim3 grid((per + 255) / 256, batch, 1);
        transpose_convert_f16<<<grid, 256, 0, stream>>>(in, o, R, Cl, ldin, Rout, Cout, sIn, sOut);
    };
    auto gemv = [&](const float* X, int ldx, long sX, const float* w, float* y, int n) {
        int total = Bx * n; dim3 grid((total + 3) / 4, 1, 1);
        gemv_dot<<<grid, 128, 0, stream>>>(X, ldx, sX, w, y, n, total, 512);
    };

    // init
    zero_f32<<<(256 * 1024 + 255) / 256, 256, 0, stream>>>(xsF, 256 * 1024);
    embed_kernel<<<dim3(Nn, Bx), 128, 0, stream>>>(x_ids, emb, xF);
    a_init_kernel<<<(Bx * Nn * Nn) / 256, 256, 0, stream>>>(adj, A0);

    float* Acur = A0; float* Aalt = A1;
    const long sXC = 131072L, sAA = 65536L;
    int nsz[4] = {256, 205, 164, 132};

    for (int l = 0; l < 3; ++l) {
        int n = nsz[l], k = nsz[l + 1];
        int np = pad16(n), npc = pad32(n), kp = pad16(k);
        long sBX = (long)np * 512, sBT = (long)512 * npc;
        long sBS = (long)np * npc, sBSk = (long)kp * npc;

        // ---- GAT: h = x @ conv_W ----
        conv(xF, 512, sXC, bX, n, 512, np, 512, sBX, Bx);
        tconv(conv_W + (size_t)l * Cc * Cc, Cc, 0, bW, Cc, Cc, 0, 1);
        gemm(bX, 512, sBX, bW, 512, 0, t0F, 512, sXC, n, 512, 512, np / 16, 32, Bx, nullptr, 0);
        gemv(t0F, 512, sXC, att_dst + (size_t)l * Cc, siF, n);
        gemv(t0F, 512, sXC, att_src + (size_t)l * Cc, sjF, n);
        masked_softmax_f16<<<dim3(np, Bx), 256, 0, stream>>>(Acur, siF, sjF, nullptr, n, npc, bS, sBS);
        tconv(t0F, 512, sXC, bT, n, 512, sBT, Bx);
        gemm(bS, npc, sBS, bT, npc, sBT, t1F, 512, sXC, n, 512, npc, np / 16, 32, Bx,
             conv_b + (size_t)l * Cc, 1);            // + bias, relu

        // ---- ASAP: GCN xp = Â @ (x @ gcn_W) + b ----
        { int total = Bx * n;
          rowsum_kernel<<<(total + 3) / 4, 128, 0, stream>>>(Acur, degF, n, total); }
        conv(t1F, 512, sXC, bX, n, 512, np, 512, sBX, Bx);
        tconv(gcn_W + (size_t)l * Cc * Cc, Cc, 0, bW, Cc, Cc, 0, 1);
        gemm(bX, 512, sBX, bW, 512, 0, t0F, 512, sXC, n, 512, 512, np / 16, 32, Bx, nullptr, 0);
        norm_adj_f16<<<dim3(np, Bx), 256, 0, stream>>>(Acur, degF, bS, n, npc, sBS);
        tconv(t0F, 512, sXC, bT, n, 512, sBT, Bx);
        gemm(bS, npc, sBS, bT, npc, sBT, t2F, 512, sXC, n, 512, npc, np / 16, 32, Bx,
             gcn_b + (size_t)l * Cc, 0);

        // ---- Xq = neighbor-max(xp); Mq = Xq @ q_W + q_b ----
        neighbor_max_kernel<<<dim3(n, Bx), 128, 0, stream>>>(Acur, t2F, t0F, n);
        conv(t0F, 512, sXC, bX, n, 512, np, 512, sBX, Bx);
        tconv(q_W + (size_t)l * Cc * Cc, Cc, 0, bW, Cc, Cc, 0, 1);
        gemm(bX, 512, sBX, bW, 512, 0, t3F, 512, sXC, n, 512, 512, np / 16, 32, Bx,
             q_b + (size_t)l * Cc, 0);

        // ---- S softmax, xc = S @ x ----
        gemv(t3F, 512, sXC, att_w + (size_t)l * 2 * Cc, siF, n);
        gemv(t2F, 512, sXC, att_w + (size_t)l * 2 * Cc + Cc, sjF, n);
        masked_softmax_f16<<<dim3(np, Bx), 256, 0, stream>>>(Acur, siF, sjF, att_b + l, n, npc, bS, sBS);
        tconv(t1F, 512, sXC, bT, n, 512, sBT, Bx);
        gemm(bS, npc, sBS, bT, npc, sBT, t3F, 512, sXC, n, 512, npc, np / 16, 32, Bx, nullptr, 0);

        // ---- LEConv fitness + top-k ----
        gemv(t3F, 512, sXC, le_W1 + (size_t)l * Cc, v1F, n);
        gemv(t3F, 512, sXC, le_W2 + (size_t)l * Cc, v2F, n);
        gemv(t3F, 512, sXC, le_W3 + (size_t)l * Cc, v3F, n);
        fit_kernel<<<dim3(n, Bx), 256, 0, stream>>>(Acur, v1F, v2F, v3F, degF, le_b1 + l, fitF, n);
        topk_kernel<<<Bx, 256, 0, stream>>>(fitF, n, k, permI, valsF);
        gather_scale_kernel<<<dim3(k, Bx), 128, 0, stream>>>(t3F, permI, valsF, xF);

        // ---- A_new = (Sk A) Sk^T + diag fix ----
        gather_rows_f16<<<dim3(kp, Bx), 256, 0, stream>>>(bS, permI, bSk, k, npc, sBS, sBSk);
        conv(Acur, 256, sAA, bAb, n, n, np, npc, sBS, Bx);
        gemm(bSk, npc, sBSk, bAb, npc, sBS, SAF, 256, sAA, k, n, npc, kp / 16, np / 16, Bx, nullptr, 0);
        conv(SAF, 256, sAA, bX, k, n, kp, npc, sBSk, Bx);
        gemm(bX, npc, sBSk, bSk, npc, sBSk, Aalt, 256, sAA, k, k, npc, kp / 16, kp / 16, Bx, nullptr, 0);
        diag_fix_kernel<<<Bx, 256, 0, stream>>>(Aalt, k);

        // ---- readout ----
        readout_kernel<<<Bx, 256, 0, stream>>>(xF, k, xsF);

        float* tmp = Acur; Acur = Aalt; Aalt = tmp;
    }

    // ---- final MLP: relu(xs @ lin1 + b) @ lin2 + b ----
    conv(xsF, 1024, 0, bX, 256, 1024, 256, 1024, 0, 1);
    tconv(lin1_W, 512, 0, bW, 1024, 512, 0, 1);
    gemm(bX, 1024, 0, bW, 1024, 0, h1F, 512, 0, 256, 512, 1024, 16, 32, 1, lin1_b, 1);
    conv(h1F, 512, 0, bX, 256, 512, 256, 512, 0, 1);
    tconv(lin2_W, 511, 0, bW, 512, 511, 0, 1);
    gemm(bX, 512, 0, bW, 512, 0, out, 511, 0, 256, 511, 512, 16, 32, 1, lin2_b, 0);
}
